// Network_21028159881518
// MI455X (gfx1250) — matrix operand order repaired and checked by reading, not compiled, not run on hardware
//
#include <hip/hip_runtime.h>

// ---------------------------------------------------------------------------
// 3D Swin transformer for MI455X (gfx1250): bf16 WMMA 16x16x32, f32 accumulate.
// All GEMMs oriented so D-fragments store contiguously (b128 LDS stores /
// float4 global RMW). MLP weights staged to LDS with
// global_load_async_to_lds_b128 (ASYNCcnt). f32->bf16 packing via
// v_cvt_pk_bf16_f32 when the builtin is available.
// ---------------------------------------------------------------------------

#define EPSV 1.1920929e-07f

typedef __attribute__((ext_vector_type(16))) __bf16 v16bf;
typedef __attribute__((ext_vector_type(2)))  __bf16 v2bf;
typedef __attribute__((ext_vector_type(8)))  float  v8f;

#if defined(__has_builtin)
#if __has_builtin(__builtin_amdgcn_cvt_pk_bf16_f32)
#define HAVE_CVT_PK_BF16 1
#endif
#endif

union FragAB { v16bf v; uint4 u[2]; };
union FragC  { v8f   v; float f[8]; };

__device__ __forceinline__ unsigned short f2bf(float f) {
    unsigned u = __float_as_uint(f);
    u += 0x7FFFu + ((u >> 16) & 1u);          // round-to-nearest-even
    return (unsigned short)(u >> 16);
}

// Pack two f32 into one dword of two bf16 (RTNE).
__device__ __forceinline__ unsigned pack2bf(float a, float b) {
#ifdef HAVE_CVT_PK_BF16
    union { v2bf v; unsigned u; } cv;
    cv.v = __builtin_amdgcn_cvt_pk_bf16_f32(a, b);
    return cv.u;
#else
    return (unsigned)f2bf(a) | ((unsigned)f2bf(b) << 16);
#endif
}

__device__ __forceinline__ uint4 pack_bf16x8(const float* f) {
    uint4 pk;
    unsigned* pw = (unsigned*)&pk;
#pragma unroll
    for (int j = 0; j < 4; ++j) pw[j] = pack2bf(f[2 * j], f[2 * j + 1]);
    return pk;
}

// A fragment: 16x32 bf16 tile from row-major [row][k] storage.
// ISA layout: lane L -> row (L&15); K chunks {c..c+7, c+16..c+23}, c=(L>>4)*8.
__device__ __forceinline__ v16bf load_a(const unsigned short* base, int row0,
                                        int kbase, int stride) {
    const int lane = threadIdx.x & 31;
    const int c    = (lane >> 4) * 8;
    const unsigned short* p = base + (row0 + (lane & 15)) * stride + kbase + c;
    FragAB t;
    t.u[0] = *(const uint4*)(p);
    t.u[1] = *(const uint4*)(p + 16);
    return t.v;
}

// B fragment: 32x16 tile from B^T stored row-major as [N][K].
// ISA layout: lane L -> col (L&15); K chunks {c..c+7, c+16..c+23}, c=(L>>4)*8.
__device__ __forceinline__ v16bf load_b(const unsigned short* bT, int col0,
                                        int kbase, int stride) {
    const int lane = threadIdx.x & 31;
    const unsigned short* p =
        bT + (col0 + (lane & 15)) * stride + kbase + ((lane >> 4) * 8);
    FragAB t;
    t.u[0] = *(const uint4*)(p);
    t.u[1] = *(const uint4*)(p + 16);
    return t.v;
}

__device__ __forceinline__ v8f wmma_bf16(v16bf a, v16bf b, v8f c) {
    return __builtin_amdgcn_wmma_f32_16x16x32_bf16(false, a, false, b,
                                                   (short)0, c, false, false);
}

__device__ __forceinline__ v8f zero_v8f() {
    FragC z;
#pragma unroll
    for (int i = 0; i < 8; ++i) z.f[i] = 0.0f;
    return z.v;
}

// Async global -> LDS copy of one 16B chunk (tracked by ASYNCcnt).
__device__ __forceinline__ void async_copy16(unsigned lds_byte_off,
                                             const void* gptr) {
    unsigned long long ga = (unsigned long long)(uintptr_t)gptr;
    asm volatile("global_load_async_to_lds_b128 %0, %1, off"
                 :: "v"(lds_byte_off), "v"(ga)
                 : "memory");
}

__device__ __forceinline__ void wait_async0() {
    asm volatile("s_wait_asynccnt 0x0" ::: "memory");
}

// ---------------------------------------------------------------------------
// Weight prep: f32 [layers][rows][cols] -> bf16 transposed [layers][cols][rows]
// ---------------------------------------------------------------------------
__global__ void transpose_bf16(const float* __restrict__ src,
                               unsigned short* __restrict__ dst,
                               int layers, int rows, int cols) {
    long total = (long)layers * rows * cols;
    for (long i = (long)blockIdx.x * blockDim.x + threadIdx.x; i < total;
         i += (long)gridDim.x * blockDim.x) {
        int l   = (int)(i / ((long)rows * cols));
        int rem = (int)(i % ((long)rows * cols));
        int r   = rem / cols;
        int c   = rem % cols;
        dst[(long)l * rows * cols + (long)c * rows + r] = f2bf(src[i]);
    }
}

// ---------------------------------------------------------------------------
// Patch embed: x(1,1,64,256,256) -> h[L=131072][64]
// ---------------------------------------------------------------------------
__global__ __launch_bounds__(64) void patch_embed(const float* __restrict__ x,
                                                  const float* __restrict__ up_w,
                                                  const float* __restrict__ up_b,
                                                  float* __restrict__ h) {
    __shared__ __align__(16) float patch[32];
    const int t   = blockIdx.x;                 // token id (gd,gh,gw)
    const int gw  = t & 63, gh = (t >> 6) & 63, gd = t >> 12;
    const int tid = threadIdx.x;
    if (tid < 32) {
        int pd2 = tid >> 4, ph2 = (tid >> 2) & 3, pw2 = tid & 3;
        long xi = ((long)(gd * 2 + pd2) * 256 + (gh * 4 + ph2)) * 256 +
                  (gw * 4 + pw2);
        patch[tid] = x[xi];
    }
    __syncthreads();
    float acc = up_b[tid];
#pragma unroll
    for (int p = 0; p < 32; ++p) acc += patch[p] * up_w[p * 64 + tid];
    h[(long)t * 64 + tid] = acc;
}

// ---------------------------------------------------------------------------
// Window attention layer (one 4^3 window per block, 4 waves, WMMA GEMMs)
// ---------------------------------------------------------------------------
__global__ __launch_bounds__(128) void attn_layer(
    float* __restrict__ h,
    const unsigned short* __restrict__ qkvWT,   // [192][64] bf16
    const unsigned short* __restrict__ projWT,  // [64][64]  bf16
    const float* __restrict__ ln1,
    const float* __restrict__ projb,
    int shift) {
    __shared__ __align__(16) unsigned short sO[64 * 64];  // norm in / attn out
    __shared__ __align__(16) unsigned short sQ[64 * 64];  // [tok][dim], scaled
    __shared__ __align__(16) unsigned short sK[64 * 64];  // [tok][dim]
    __shared__ __align__(16) unsigned short sVT[64 * 64]; // [dim][tok]
    __shared__ __align__(16) float          sS[64 * 64];  // per-head [q][k]
    __shared__ __align__(16) unsigned short sP[64 * 64];  // per-head [q][k]
    __shared__ int   otok[64];
    __shared__ int   rid[64];
    __shared__ float red[128];

    const int tid    = threadIdx.x;
    const int wv     = tid >> 5;
    const int lane   = tid & 31;
    const int half16 = lane >> 4;
    const int col    = lane & 15;

    const int wid = blockIdx.x;
    const int ww = wid & 15, wh = (wid >> 4) & 15, wd = wid >> 8;

    if (tid < 64) {
        int rd = tid >> 4, rh = (tid >> 2) & 3, rw = tid & 3;
        int gd = wd * 4 + rd, gh = wh * 4 + rh, gw = ww * 4 + rw;
        int od = (gd + shift) & 31;
        int oh = (gh + shift) & 63;
        int ow = (gw + shift) & 63;
        otok[tid] = (od << 12) + (oh << 6) + ow;
        int r0 = (gd < 28) ? 0 : ((gd < 30) ? 1 : 2);
        int r1 = (gh < 60) ? 0 : ((gh < 62) ? 1 : 2);
        int r2 = (gw < 60) ? 0 : ((gw < 62) ? 1 : 2);
        rid[tid] = r0 * 9 + r1 * 3 + r2;
    }
    __syncthreads();

    // ---- RMSNorm -> bf16 into sO [token][dim] ----------------------------
    {
        const int t = tid & 63, hf = tid >> 6;
        const float* row = h + (long)otok[t] * 64 + hf * 32;
        float v[32];
        float ss = 0.0f;
#pragma unroll
        for (int i = 0; i < 32; ++i) { v[i] = row[i]; ss += v[i] * v[i]; }
        red[tid] = ss;
        __syncthreads();
        float sc = rsqrtf((red[t] + red[t + 64]) * (1.0f / 64.0f) + EPSV);
        float nv[8];
#pragma unroll
        for (int b = 0; b < 4; ++b) {
#pragma unroll
            for (int j = 0; j < 8; ++j)
                nv[j] = v[b * 8 + j] * sc * ln1[hf * 32 + b * 8 + j];
            *(uint4*)&sO[t * 64 + hf * 32 + b * 8] = pack_bf16x8(nv);
        }
    }
    __syncthreads();

    // ---- QKV: Q^T,K^T orientation (A=weights) ; V normal -> sVT ----------
    const float qscale = 0.17677669529663687f;  // 1/sqrt(32)
    for (int tile = wv; tile < 48; tile += 4) {
        int tu = __builtin_amdgcn_readfirstlane(tile);
        if (tu < 32) {
            // D[M=dim][N=token] ; dims 0..63 = Q, 64..127 = K
            int mt = tile >> 2, nt = tile & 3;
            FragC acc; acc.v = zero_v8f();
#pragma unroll
            for (int k = 0; k < 64; k += 32)
                acc.v = wmma_bf16(load_a(qkvWT, mt * 16, k, 64),
                                  load_b(sO, nt * 16, k, 64), acc.v);
            int tok = nt * 16 + col;
            int mb  = (mt & 3) * 16 + half16 * 8;
            if (tu < 16) {                   // Q (pre-scaled)
                float tmp[8];
#pragma unroll
                for (int r = 0; r < 8; ++r) tmp[r] = acc.f[r] * qscale;
                *(uint4*)&sQ[tok * 64 + mb] = pack_bf16x8(tmp);
            } else {                         // K
                *(uint4*)&sK[tok * 64 + mb] = pack_bf16x8(acc.f);
            }
        } else {
            // V: D[M=token][N=dim] -> store transposed into sVT[dim][token]
            int t2 = tile - 32;
            int mt = t2 >> 2, nt = t2 & 3;
            FragC acc; acc.v = zero_v8f();
#pragma unroll
            for (int k = 0; k < 64; k += 32)
                acc.v = wmma_bf16(load_a(sO, mt * 16, k, 64),
                                  load_b(qkvWT, 128 + nt * 16, k, 64), acc.v);
            int d  = nt * 16 + col;
            int mb = mt * 16 + half16 * 8;
            *(uint4*)&sVT[d * 64 + mb] = pack_bf16x8(acc.f);
        }
    }
    __syncthreads();

    // ---- per-head: scores, softmax, P@V ----------------------------------
    for (int hh = 0; hh < 2; ++hh) {
        // S^T = K @ Q^T : D[M=k-tok][N=q-tok] -> sS[q][k] (contiguous in k)
        for (int tile = wv; tile < 16; tile += 4) {
            int mt = tile >> 2, nt = tile & 3;
            FragC acc; acc.v = zero_v8f();
            acc.v = wmma_bf16(load_a(sK, mt * 16, hh * 32, 64),
                              load_b(sQ, nt * 16, hh * 32, 64), acc.v);
            int q  = nt * 16 + col;
            int mb = mt * 16 + half16 * 8;
            int rq = rid[q];
            float4 v0, v1;
            float* o0 = (float*)&v0;
            float* o1 = (float*)&v1;
#pragma unroll
            for (int r = 0; r < 4; ++r) {
                float p0 = (rid[mb + r] != rq) ? -100.0f : 0.0f;
                float p1 = (rid[mb + 4 + r] != rq) ? -100.0f : 0.0f;
                o0[r] = acc.f[r]     + (shift ? p0 : 0.0f);
                o1[r] = acc.f[4 + r] + (shift ? p1 : 0.0f);
            }
            *(float4*)&sS[q * 64 + mb]     = v0;
            *(float4*)&sS[q * 64 + mb + 4] = v1;
        }
        __syncthreads();

        // softmax rows -> bf16 probs sP[q][k]
        if (tid < 64) {
            float* row = sS + tid * 64;
            float mx = -1e30f;
            for (int i = 0; i < 64; ++i) mx = fmaxf(mx, row[i]);
            float sum = 0.0f;
            for (int i = 0; i < 64; ++i) {
                float e = __expf(row[i] - mx);
                row[i] = e;
                sum += e;
            }
            float inv = 1.0f / sum;
            for (int i = 0; i < 64; ++i) sP[tid * 64 + i] = f2bf(row[i] * inv);
        }
        __syncthreads();

        // O^T = V^T @ P^T : D[M=dim][N=q-tok] -> sO[tok][dim] (contiguous)
        for (int tile = wv; tile < 8; tile += 4) {
            int mt = tile >> 2, nt = tile & 3;
            FragC acc; acc.v = zero_v8f();
#pragma unroll
            for (int k = 0; k < 64; k += 32)
                acc.v = wmma_bf16(load_a(sVT, hh * 32 + mt * 16, k, 64),
                                  load_b(sP, nt * 16, k, 64), acc.v);
            int q  = nt * 16 + col;
            int mb = hh * 32 + mt * 16 + half16 * 8;
            *(uint4*)&sO[q * 64 + mb] = pack_bf16x8(acc.f);
        }
        __syncthreads();
    }

    // ---- proj: D[M=ch][N=token] -> vectorized global RMW -----------------
    for (int tile = wv; tile < 16; tile += 4) {
        int mt = tile >> 2, nt = tile & 3;
        FragC acc; acc.v = zero_v8f();
#pragma unroll
        for (int k = 0; k < 64; k += 32)
            acc.v = wmma_bf16(load_a(projWT, mt * 16, k, 64),
                              load_b(sO, nt * 16, k, 64), acc.v);
        int tok = nt * 16 + col;
        int mb  = mt * 16 + half16 * 8;
        long gb = (long)otok[tok] * 64 + mb;
        float4 h0  = *(const float4*)&h[gb];
        float4 h1  = *(const float4*)&h[gb + 4];
        float4 pb0 = *(const float4*)&projb[mb];
        float4 pb1 = *(const float4*)&projb[mb + 4];
        h0.x += acc.f[0] + pb0.x; h0.y += acc.f[1] + pb0.y;
        h0.z += acc.f[2] + pb0.z; h0.w += acc.f[3] + pb0.w;
        h1.x += acc.f[4] + pb1.x; h1.y += acc.f[5] + pb1.y;
        h1.z += acc.f[6] + pb1.z; h1.w += acc.f[7] + pb1.w;
        *(float4*)&h[gb]     = h0;
        *(float4*)&h[gb + 4] = h1;
    }
}

// ---------------------------------------------------------------------------
// MLP layer: RMSNorm -> W1+GELU -> W2 + residual (64 tokens per block).
// Weights staged to LDS with async copies overlapping the RMSNorm.
// ---------------------------------------------------------------------------
__global__ __launch_bounds__(128) void mlp_layer(
    float* __restrict__ h,
    const unsigned short* __restrict__ w1T,  // [128][64] bf16 (16KB)
    const unsigned short* __restrict__ w2T,  // [64][128] bf16 (16KB)
    const float* __restrict__ ln2,
    const float* __restrict__ b1,
    const float* __restrict__ b2) {
    __shared__ __align__(16) unsigned short sA[64 * 64];    // [tok][dim]
    __shared__ __align__(16) unsigned short sH[64 * 128];   // [tok][hid]
    __shared__ __align__(16) unsigned short sW1[128 * 64];
    __shared__ __align__(16) unsigned short sW2[64 * 128];
    __shared__ float red[128];

    const int tid    = threadIdx.x;
    const int wv     = tid >> 5;
    const int lane   = tid & 31;
    const int half16 = lane >> 4;
    const int col    = lane & 15;
    const long base  = (long)blockIdx.x * 64;

    // ---- kick off async weight staging (32KB total, 2048 x b128) ---------
    {
        const unsigned l1 = (unsigned)(uintptr_t)(&sW1[0]);
        const unsigned l2 = (unsigned)(uintptr_t)(&sW2[0]);
        const char* g1 = (const char*)w1T;
        const char* g2 = (const char*)w2T;
#pragma unroll
        for (int i = 0; i < 8; ++i) {
            int off = (i * 128 + tid) * 16;
            async_copy16(l1 + off, g1 + off);
            async_copy16(l2 + off, g2 + off);
        }
    }

    // ---- RMSNorm (overlaps the async copies) -----------------------------
    {
        const int t = tid & 63, hf = tid >> 6;
        const float* row = h + (base + t) * 64 + hf * 32;
        float v[32];
        float ss = 0.0f;
#pragma unroll
        for (int i = 0; i < 32; ++i) { v[i] = row[i]; ss += v[i] * v[i]; }
        red[tid] = ss;
        __syncthreads();
        float sc = rsqrtf((red[t] + red[t + 64]) * (1.0f / 64.0f) + EPSV);
        float nv[8];
#pragma unroll
        for (int b = 0; b < 4; ++b) {
#pragma unroll
            for (int j = 0; j < 8; ++j)
                nv[j] = v[b * 8 + j] * sc * ln2[hf * 32 + b * 8 + j];
            *(uint4*)&sA[t * 64 + hf * 32 + b * 8] = pack_bf16x8(nv);
        }
    }
    wait_async0();
    __syncthreads();

    // H1^T = W1 @ A^T : D[M=hid][N=tok] -> gelu -> sH[tok][hid] (contiguous)
    for (int tile = wv; tile < 32; tile += 4) {
        int mt = tile >> 2, nt = tile & 3;       // mt: hidden tile 0..7
        FragC acc; acc.v = zero_v8f();
#pragma unroll
        for (int k = 0; k < 64; k += 32)
            acc.v = wmma_bf16(load_a(sW1, mt * 16, k, 64),
                              load_b(sA, nt * 16, k, 64), acc.v);
        int tok = nt * 16 + col;
        int mb  = mt * 16 + half16 * 8;
        float tmp[8];
#pragma unroll
        for (int r = 0; r < 8; ++r) {
            float xx = acc.f[r] + b1[mb + r];
            tmp[r] = 0.5f * xx * (1.0f + erff(xx * 0.70710678118654752f));
        }
        *(uint4*)&sH[tok * 128 + mb] = pack_bf16x8(tmp);
    }
    __syncthreads();

    // out^T = W2 @ H1^T : D[M=ch][N=tok] -> vectorized global RMW ----------
    for (int tile = wv; tile < 16; tile += 4) {
        int mt = tile >> 2, nt = tile & 3;
        FragC acc; acc.v = zero_v8f();
#pragma unroll
        for (int k = 0; k < 128; k += 32)
            acc.v = wmma_bf16(load_a(sW2, mt * 16, k, 128),
                              load_b(sH, nt * 16, k, 128), acc.v);
        int tok = nt * 16 + col;
        int mb  = mt * 16 + half16 * 8;
        long gb = (base + tok) * 64 + mb;
        float4 h0  = *(const float4*)&h[gb];
        float4 h1  = *(const float4*)&h[gb + 4];
        float4 bb0 = *(const float4*)&b2[mb];
        float4 bb1 = *(const float4*)&b2[mb + 4];
        h0.x += acc.f[0] + bb0.x; h0.y += acc.f[1] + bb0.y;
        h0.z += acc.f[2] + bb0.z; h0.w += acc.f[3] + bb0.w;
        h1.x += acc.f[4] + bb1.x; h1.y += acc.f[5] + bb1.y;
        h1.z += acc.f[6] + bb1.z; h1.w += acc.f[7] + bb1.w;
        *(float4*)&h[gb]     = h0;
        *(float4*)&h[gb + 4] = h1;
    }
}

// ---------------------------------------------------------------------------
// Down-projections + fold back to voxels
// ---------------------------------------------------------------------------
__global__ __launch_bounds__(64) void down_fold(const float* __restrict__ h,
                                                const float* __restrict__ pw,
                                                const float* __restrict__ pb,
                                                const float* __restrict__ cw,
                                                const float* __restrict__ cb,
                                                float* __restrict__ out) {
    __shared__ __align__(16) float row[64];
    const int t   = blockIdx.x;
    const int tid = threadIdx.x;
    row[tid] = h[(long)t * 64 + tid];
    __syncthreads();

    const int j = tid & 31;
    const float* w = (tid < 32) ? pw : cw;
    float acc      = (tid < 32) ? pb[j] : cb[j];
#pragma unroll
    for (int c = 0; c < 64; ++c) acc += row[c] * w[c * 32 + j];

    const int gw = t & 63, gh = (t >> 6) & 63, gd = t >> 12;
    const int pd2 = j >> 4, ph2 = (j >> 2) & 3, pw2 = j & 3;
    long vox = ((long)(gd * 2 + pd2) * 256 + (gh * 4 + ph2)) * 256 +
               (gw * 4 + pw2);
    out[((tid < 32) ? 0L : 4194304L) + vox] = acc;
}

// ---------------------------------------------------------------------------
extern "C" void kernel_launch(void* const* d_in, const int* in_sizes, int n_in,
                              void* d_out, int out_size, void* d_ws,
                              size_t ws_size, hipStream_t stream) {
    (void)in_sizes; (void)n_in; (void)out_size; (void)ws_size;

    const float* x      = (const float*)d_in[0];
    const float* up_w   = (const float*)d_in[1];
    const float* up_b   = (const float*)d_in[2];
    const float* ln1_w  = (const float*)d_in[3];
    const float* qkv_w  = (const float*)d_in[4];
    const float* proj_w = (const float*)d_in[5];
    const float* proj_b = (const float*)d_in[6];
    const float* ln2_w  = (const float*)d_in[7];
    const float* mlp_w1 = (const float*)d_in[8];
    const float* mlp_b1 = (const float*)d_in[9];
    const float* mlp_w2 = (const float*)d_in[10];
    const float* mlp_b2 = (const float*)d_in[11];
    const float* dpw    = (const float*)d_in[12];
    const float* dpb    = (const float*)d_in[13];
    const float* dcw    = (const float*)d_in[14];
    const float* dcb    = (const float*)d_in[15];
    float* out = (float*)d_out;

    char* ws = (char*)d_ws;
    float*          h      = (float*)(ws);                      // 33,554,432 B
    unsigned short* qkvWT  = (unsigned short*)(ws + 33554432);  //    393,216 B
    unsigned short* projWT = (unsigned short*)(ws + 33947648);  //    131,072 B
    unsigned short* w1T    = (unsigned short*)(ws + 34078720);  //    262,144 B
    unsigned short* w2T    = (unsigned short*)(ws + 34340864);  //    262,144 B

    // weight prep (bf16, transposed)
    transpose_bf16<<<256, 256, 0, stream>>>(qkv_w, qkvWT, 16, 64, 192);
    transpose_bf16<<<256, 256, 0, stream>>>(proj_w, projWT, 16, 64, 64);
    transpose_bf16<<<256, 256, 0, stream>>>(mlp_w1, w1T, 16, 64, 128);
    transpose_bf16<<<256, 256, 0, stream>>>(mlp_w2, w2T, 16, 128, 64);

    patch_embed<<<131072, 64, 0, stream>>>(x, up_w, up_b, h);

    for (int l = 0; l < 16; ++l) {
        int shift = (l & 1) ? 2 : 0;
        attn_layer<<<2048, 128, 0, stream>>>(
            h, qkvWT + (size_t)l * 192 * 64, projWT + (size_t)l * 64 * 64,
            ln1_w + l * 64, proj_b + l * 64, shift);
        mlp_layer<<<2048, 128, 0, stream>>>(
            h, w1T + (size_t)l * 128 * 64, w2T + (size_t)l * 64 * 128,
            ln2_w + l * 64, mlp_b1 + l * 128, mlp_b2 + l * 64);
    }

    down_fold<<<131072, 64, 0, stream>>>(h, dpw, dpb, dcw, dcb, out);
}
